// ChineseCLIPVisionAttentionWrapper_66348654789025
// MI455X (gfx1250) — matrix operand order repaired and checked by reading, not compiled, not run on hardware
//
#include <hip/hip_runtime.h>

typedef __attribute__((ext_vector_type(16))) _Float16 v16h;
typedef __attribute__((ext_vector_type(8)))  float    v8f;

#define T_TOK 257
#define BATCH 32
#define EMBED 1024
#define HEADS 16
#define HD    64
#define TP    288                 // padded sequence length (multiple of 32)
#define MROWS (BATCH * T_TOK)     // 8224 = 32 * 257
#define QSCALE 0.125f             // 64^-0.5

// ---------------------------------------------------------------------------
// WMMA fragment loaders (16-bit data, 16x16x32 shape, wave32 layouts from ISA)
//
// A (16x32, row-major source):
//   lane l: row = l&15; g = l>>4
//   halves: K in [g*8, g*8+8) and [16+g*8, 16+g*8+8)  -> two 16B loads
// B (32x16, column n contiguous in memory, i.e. "row n of a row-major W"):
//   lane l: col = l&15; g = l>>4
//   halves: K in [g*8, g*8+8) and [16+g*8, 16+g*8+8)  -> two 16B loads
// ---------------------------------------------------------------------------
__device__ inline v16h load_a16(const _Float16* A, int lda, int row0, int k0, int lane) {
    const int m = lane & 15, g = lane >> 4;
    const _Float16* p = A + (size_t)(row0 + m) * lda + (k0 + g * 8);
    union { v16h v; float4 q[2]; } u;
    u.q[0] = *(const float4*)(p);
    u.q[1] = *(const float4*)(p + 16);
    return u.v;
}

__device__ inline v16h load_b16(const _Float16* Bc, int ldb, int col0, int k0, int lane) {
    const int n = lane & 15, g = lane >> 4;
    const _Float16* p = Bc + (size_t)(col0 + n) * ldb + (k0 + g * 8);
    union { v16h v; float4 q[2]; } u;
    u.q[0] = *(const float4*)(p);
    u.q[1] = *(const float4*)(p + 16);
    return u.v;
}

__device__ inline v8f wmma16(v16h a, v16h b, v8f c) {
    return __builtin_amdgcn_wmma_f32_16x16x32_f16(false, a, false, b, (short)0, c, false, false);
}

// ---------------------------------------------------------------------------
// Elementwise helpers
// ---------------------------------------------------------------------------
__global__ void cast_f32_f16(const float* __restrict__ src, _Float16* __restrict__ dst, int n) {
    int i = blockIdx.x * blockDim.x + threadIdx.x;
    if (i < n) dst[i] = (_Float16)src[i];
}

__global__ void zero_f4(float4* __restrict__ p, int n) {
    int i = blockIdx.x * blockDim.x + threadIdx.x;
    if (i < n) p[i] = make_float4(0.f, 0.f, 0.f, 0.f);
}

// ---------------------------------------------------------------------------
// QKV projection: out[m][n] = sum_k X[m][k] * W[n][k] + bias[n]
// One wave per 32x64 output tile (B fragments reused across two row tiles).
// z-dim selects Q/K/V; Q scaled, V written transposed.
// __launch_bounds__(32, 1): single-wave blocks, max VGPR budget (no spills).
// ---------------------------------------------------------------------------
__global__ __launch_bounds__(32, 1)
void qkv_gemm(const _Float16* __restrict__ X,
              const _Float16* __restrict__ Wq, const _Float16* __restrict__ Wk,
              const _Float16* __restrict__ Wv,
              const float* __restrict__ bq, const float* __restrict__ bk,
              const float* __restrict__ bv,
              _Float16* __restrict__ Q, _Float16* __restrict__ K,
              _Float16* __restrict__ Vt) {
    const int lane = threadIdx.x;
    const int row0 = blockIdx.x * 32;
    const int col0 = blockIdx.y * 64;
    const int mat  = blockIdx.z;
    const _Float16* W   = (mat == 0) ? Wq : (mat == 1) ? Wk : Wv;
    const float*    bia = (mat == 0) ? bq : (mat == 1) ? bk : bv;

    v8f acc[2][4] = {};
    for (int kc = 0; kc < EMBED / 32; ++kc) {
        v16h a0 = load_a16(X, EMBED, row0,      kc * 32, lane);
        v16h a1 = load_a16(X, EMBED, row0 + 16, kc * 32, lane);
#pragma unroll
        for (int nt = 0; nt < 4; ++nt) {
            v16h b = load_b16(W, EMBED, col0 + nt * 16, kc * 32, lane);
            acc[0][nt] = wmma16(a0, b, acc[0][nt]);
            acc[1][nt] = wmma16(a1, b, acc[1][nt]);
        }
    }

    const int g = lane >> 4, nl = lane & 15;
#pragma unroll
    for (int rt = 0; rt < 2; ++rt) {
#pragma unroll
        for (int nt = 0; nt < 4; ++nt) {
            const int   n  = col0 + nt * 16 + nl;
            const int   h  = n >> 6, d = n & 63;
            const float bb = bia[n];
#pragma unroll
            for (int r = 0; r < 8; ++r) {
                const int m   = row0 + rt * 16 + r + 8 * g;
                const int b_  = m / T_TOK;
                const int tok = m % T_TOK;
                const float v = acc[rt][nt][r] + bb;
                if (mat == 0) {
                    Q[(((size_t)(b_ * HEADS + h)) * TP + tok) * HD + d] = (_Float16)(v * QSCALE);
                } else if (mat == 1) {
                    K[(((size_t)(b_ * HEADS + h)) * TP + tok) * HD + d] = (_Float16)v;
                } else {
                    Vt[(((size_t)(b_ * HEADS + h)) * HD + d) * TP + tok] = (_Float16)v;
                }
            }
        }
    }
}

// ---------------------------------------------------------------------------
// Attention: one wave per (batch*head, 16-row q tile).
//   S = Q Kt (2 WMMA per key tile) -> LDS f32 panel (masked past 257)
//   softmax across 288 cols (2 lanes per row, shfl-combined)
//   ctx = P V (A from LDS f16 panel, B from transposed V in global)
// ---------------------------------------------------------------------------
__global__ __launch_bounds__(32, 1)
void attn_kernel(const _Float16* __restrict__ Q, const _Float16* __restrict__ Km,
                 const _Float16* __restrict__ Vt, _Float16* __restrict__ ctx) {
    __shared__ __attribute__((aligned(16))) float    S[16][TP];
    __shared__ __attribute__((aligned(16))) _Float16 P[16][TP];

    const int lane = threadIdx.x;
    const int bh   = blockIdx.x;     // 0..511
    const int qt   = blockIdx.y;     // 0..16
    const int g = lane >> 4, nl = lane & 15;

    const _Float16* Qb = Q  + (size_t)bh * TP * HD;
    const _Float16* Kb = Km + (size_t)bh * TP * HD;
    const _Float16* Vb = Vt + (size_t)bh * HD * TP;

    // Q fragments: hd=64 -> two k32 chunks, reused across all 18 key tiles
    const v16h qa0 = load_a16(Qb, HD, qt * 16, 0,  lane);
    const v16h qa1 = load_a16(Qb, HD, qt * 16, 32, lane);

    for (int j = 0; j < TP / 16; ++j) {
        v16h kb0 = load_b16(Kb, HD, j * 16, 0,  lane);
        v16h kb1 = load_b16(Kb, HD, j * 16, 32, lane);
        v8f s = {};
        s = wmma16(qa0, kb0, s);
        s = wmma16(qa1, kb1, s);
        const int  col    = j * 16 + nl;
        const bool masked = (col >= T_TOK);
#pragma unroll
        for (int r = 0; r < 8; ++r)
            S[r + 8 * g][col] = masked ? -1e30f : s[r];
    }
    __syncthreads();

    // softmax: row = nl, lane-half g owns cols [g*144, g*144+144)
    {
        const int row = nl;
        const int c0  = g * (TP / 2);
        float mx = -1e30f;
        for (int c = 0; c < TP / 2; ++c) mx = fmaxf(mx, S[row][c0 + c]);
        mx = fmaxf(mx, __shfl_xor(mx, 16, 32));
        float sum = 0.f;
        for (int c = 0; c < TP / 2; ++c) {
            float e = __expf(S[row][c0 + c] - mx);
            S[row][c0 + c] = e;
            sum += e;
        }
        sum += __shfl_xor(sum, 16, 32);
        const float inv = 1.0f / sum;
        for (int c = 0; c < TP / 2; ++c)
            P[row][c0 + c] = (_Float16)(S[row][c0 + c] * inv);
    }
    __syncthreads();

    // ctx = P (16 x 288) x V (288 x 64)
    const int b_ = bh >> 4, h = bh & 15;
#pragma unroll
    for (int nt = 0; nt < 4; ++nt) {
        v8f acc = {};
        for (int c = 0; c < TP / 32; ++c) {
            v16h pa = load_a16(&P[0][0], TP, 0, c * 32, lane);
            v16h vb = load_b16(Vb, TP, nt * 16, c * 32, lane);
            acc = wmma16(pa, vb, acc);
        }
        const int d = nt * 16 + nl;
#pragma unroll
        for (int r = 0; r < 8; ++r) {
            const int tok = qt * 16 + r + 8 * g;
            if (tok < T_TOK)
                ctx[((size_t)b_ * T_TOK + tok) * EMBED + h * HD + d] = (_Float16)acc[r];
        }
    }
}

// ---------------------------------------------------------------------------
// Output projection: out[m][n] = sum_k CTX[m][k] * Wo[n][k] + bo[n]  (f32 out)
// One wave per 32x64 output tile.
// ---------------------------------------------------------------------------
__global__ __launch_bounds__(32, 1)
void out_gemm(const _Float16* __restrict__ X, const _Float16* __restrict__ Wo,
              const float* __restrict__ bo, float* __restrict__ out) {
    const int lane = threadIdx.x;
    const int row0 = blockIdx.x * 32;
    const int col0 = blockIdx.y * 64;

    v8f acc[2][4] = {};
    for (int kc = 0; kc < EMBED / 32; ++kc) {
        v16h a0 = load_a16(X, EMBED, row0,      kc * 32, lane);
        v16h a1 = load_a16(X, EMBED, row0 + 16, kc * 32, lane);
#pragma unroll
        for (int nt = 0; nt < 4; ++nt) {
            v16h b = load_b16(Wo, EMBED, col0 + nt * 16, kc * 32, lane);
            acc[0][nt] = wmma16(a0, b, acc[0][nt]);
            acc[1][nt] = wmma16(a1, b, acc[1][nt]);
        }
    }
    const int g = lane >> 4, nl = lane & 15;
#pragma unroll
    for (int rt = 0; rt < 2; ++rt) {
#pragma unroll
        for (int nt = 0; nt < 4; ++nt) {
            const int   n  = col0 + nt * 16 + nl;
            const float bb = bo[n];
#pragma unroll
            for (int r = 0; r < 8; ++r) {
                const int m = row0 + rt * 16 + r + 8 * g;
                out[(size_t)m * EMBED + n] = acc[rt][nt][r] + bb;
            }
        }
    }
}

// ---------------------------------------------------------------------------
extern "C" void kernel_launch(void* const* d_in, const int* in_sizes, int n_in,
                              void* d_out, int out_size, void* d_ws, size_t ws_size,
                              hipStream_t stream) {
    (void)in_sizes; (void)n_in; (void)out_size; (void)ws_size;

    const float* hid = (const float*)d_in[0];
    const float* Wq  = (const float*)d_in[1];
    const float* bq  = (const float*)d_in[2];
    const float* Wk  = (const float*)d_in[3];
    const float* bk  = (const float*)d_in[4];
    const float* Wv  = (const float*)d_in[5];
    const float* bv  = (const float*)d_in[6];
    const float* Wo  = (const float*)d_in[7];
    const float* bo  = (const float*)d_in[8];
    float* out = (float*)d_out;

    // workspace carve-up (all chunks 16B-aligned)
    char* ws = (char*)d_ws;
    size_t off = 0;
    _Float16* X16  = (_Float16*)(ws + off); off += (size_t)MROWS * EMBED * sizeof(_Float16);
    _Float16* Wq16 = (_Float16*)(ws + off); off += (size_t)EMBED * EMBED * sizeof(_Float16);
    _Float16* Wk16 = (_Float16*)(ws + off); off += (size_t)EMBED * EMBED * sizeof(_Float16);
    _Float16* Wv16 = (_Float16*)(ws + off); off += (size_t)EMBED * EMBED * sizeof(_Float16);
    _Float16* Wo16 = (_Float16*)(ws + off); off += (size_t)EMBED * EMBED * sizeof(_Float16);
    _Float16* Qb   = (_Float16*)(ws + off); off += (size_t)BATCH * HEADS * TP * HD * sizeof(_Float16);
    _Float16* Kb   = (_Float16*)(ws + off); off += (size_t)BATCH * HEADS * TP * HD * sizeof(_Float16);
    _Float16* Vtb  = (_Float16*)(ws + off); off += (size_t)BATCH * HEADS * HD * TP * sizeof(_Float16);
    _Float16* CTX  = (_Float16*)(ws + off); off += (size_t)MROWS * EMBED * sizeof(_Float16);

    // 1) casts to f16
    const int nX = MROWS * EMBED;
    cast_f32_f16<<<(nX + 255) / 256, 256, 0, stream>>>(hid, X16, nX);
    const int nW = EMBED * EMBED;
    cast_f32_f16<<<(nW + 255) / 256, 256, 0, stream>>>(Wq, Wq16, nW);
    cast_f32_f16<<<(nW + 255) / 256, 256, 0, stream>>>(Wk, Wk16, nW);
    cast_f32_f16<<<(nW + 255) / 256, 256, 0, stream>>>(Wv, Wv16, nW);
    cast_f32_f16<<<(nW + 255) / 256, 256, 0, stream>>>(Wo, Wo16, nW);

    // 2) zero Q/K/Vt pad regions (the three buffers are contiguous)
    const size_t zBytes = 3 * (size_t)BATCH * HEADS * TP * HD * sizeof(_Float16);
    const int nZ = (int)(zBytes / 16);
    zero_f4<<<(nZ + 255) / 256, 256, 0, stream>>>((float4*)Qb, nZ);

    // 3) QKV projections (32-row x 64-col tiles per wave; 8224 = 32 * 257)
    qkv_gemm<<<dim3(MROWS / 32, EMBED / 64, 3), 32, 0, stream>>>(
        X16, Wq16, Wk16, Wv16, bq, bk, bv, Qb, Kb, Vtb);

    // 4) attention (512 bh slices x 17 q-tiles covering 257 tokens)
    attn_kernel<<<dim3(BATCH * HEADS, 17), 32, 0, stream>>>(Qb, Kb, Vtb, CTX);

    // 5) output projection
    out_gemm<<<dim3(MROWS / 32, EMBED / 64), 32, 0, stream>>>(CTX, Wo16, bo, out);
}
